// RegularizedTransformerBlock_39556648796553
// MI455X (gfx1250) — matrix operand order repaired and checked by reading, not compiled, not run on hardware
//
#include <hip/hip_runtime.h>
#include <hip/hip_bf16.h>
#include <math.h>

typedef __attribute__((ext_vector_type(16))) _Float16 v16h;
typedef __attribute__((ext_vector_type(8)))  float    v8f;
typedef __attribute__((ext_vector_type(2)))  int      v2i;

#define D_MODEL 1024
#define L_SEQ 2048
#define NTOK 8192           // 4 * 2048
#define NHEAD 16
#define HDIM 64

#if defined(__AMDGCN__) && __has_builtin(__builtin_amdgcn_global_load_async_to_lds_b64) && \
    __has_builtin(__builtin_amdgcn_s_wait_asynccnt)
#define HAVE_ASYNC 1
#else
#define HAVE_ASYNC 0
#endif

// 8-byte global->LDS copy: async DMA path on CDNA5 when available
static __device__ __forceinline__ void cp8(_Float16* l, const _Float16* g) {
#if HAVE_ASYNC
    __builtin_amdgcn_global_load_async_to_lds_b64(
        (__attribute__((address_space(1))) v2i*)(v2i*)(void*)g,
        (__attribute__((address_space(3))) v2i*)(v2i*)(void*)l, 0, 0);
#else
    *reinterpret_cast<uint2*>(l) = *reinterpret_cast<const uint2*>(g);
#endif
}

static __device__ __forceinline__ void async_fence() {
#if HAVE_ASYNC
    __builtin_amdgcn_s_wait_asynccnt(0);
#endif
}

static __device__ __forceinline__ v8f wmma_f16(v16h a, v16h b, v8f c) {
    // D = A(16x32 f16) * B(32x16 f16) + C(16x16 f32)
    return __builtin_amdgcn_wmma_f32_16x16x32_f16(
        false, a, false, b, (short)0, c, false, false);
}

// A-matrix fragment (16x32 f16), contiguous k = 16*half + s
static __device__ __forceinline__ v16h load_a_frag(const _Float16* A, int lda,
                                                   int mbase, int kbase, int lane) {
    int m = mbase + (lane & 15);
    int hlf = lane >> 4;
    v16h f;
#pragma unroll
    for (int v = 0; v < 8; ++v) {
        int k = kbase + (hlf << 4) + (v << 1);
        f[2 * v]     = A[(size_t)m * lda + k];
        f[2 * v + 1] = A[(size_t)m * lda + k + 1];
    }
    return f;
}

// B-matrix fragment (32x16 f16), row-major source [K][N]
static __device__ __forceinline__ v16h load_b_frag(const _Float16* B, int ldb,
                                                   int nbase, int lane) {
    int n = nbase + (lane & 15);
    int hlf = lane >> 4;
    v16h f;
#pragma unroll
    for (int v = 0; v < 8; ++v) {
        int k = (hlf << 4) + (v << 1);
        f[2 * v]     = B[(size_t)k * ldb + n];
        f[2 * v + 1] = B[(size_t)(k + 1) * ldb + n];
    }
    return f;
}

// B fragment for K^T: source row-major [keys][dims]; B[kdim=d][n=key j] = src[j][d]
static __device__ __forceinline__ v16h load_bT_frag(const _Float16* S, int lds,
                                                    int jbase, int dbase, int lane) {
    int j = jbase + (lane & 15);
    int hlf = lane >> 4;
    v16h f;
#pragma unroll
    for (int v = 0; v < 8; ++v) {
        int d = dbase + (hlf << 4) + (v << 1);
        f[2 * v]     = S[(size_t)j * lds + d];
        f[2 * v + 1] = S[(size_t)j * lds + d + 1];
    }
    return f;
}

// ---------------- fp32 -> fp16 convert ----------------
__global__ void cvt_k(const float* __restrict__ in, _Float16* __restrict__ out, int n) {
    int i = blockIdx.x * 256 + threadIdx.x;
    if (i < n) out[i] = (_Float16)in[i];
}

// ---------------- LayerNorm (fp32 in -> f16 out) ----------------
__global__ __launch_bounds__(256) void ln_k(const float* __restrict__ x,
                                            const float* __restrict__ g,
                                            const float* __restrict__ b,
                                            _Float16* __restrict__ out) {
    __shared__ float r1[256], r2[256];
    int row = blockIdx.x, tid = threadIdx.x;
    const float* xr = x + (size_t)row * D_MODEL;
    float s = 0.f, s2 = 0.f;
    for (int i = tid; i < D_MODEL; i += 256) { float v = xr[i]; s += v; s2 += v * v; }
    r1[tid] = s; r2[tid] = s2;
    __syncthreads();
    for (int off = 128; off > 0; off >>= 1) {
        if (tid < off) { r1[tid] += r1[tid + off]; r2[tid] += r2[tid + off]; }
        __syncthreads();
    }
    float mu = r1[0] * (1.f / D_MODEL);
    float var = r2[0] * (1.f / D_MODEL) - mu * mu;
    float rstd = rsqrtf(var + 1e-5f);
    for (int i = tid; i < D_MODEL; i += 256)
        out[(size_t)row * D_MODEL + i] = (_Float16)((xr[i] - mu) * rstd * g[i] + b[i]);
}

// ---------------- WMMA GEMM: block tile 128x256, wave tile 64x64 ------------------
// MODE 0: f16 (acc+bias); MODE 1: f16 GELU(acc+bias); MODE 2: f32 acc+bias+res
#define LDA 40    // 128 x 32 halves, padded
#define LDB 264   // 32 x 256 halves, padded
template <int MODE>
__global__ __launch_bounds__(256) void gemm_k(const _Float16* __restrict__ A,
                                              const _Float16* __restrict__ B,
                                              const float* __restrict__ bias,
                                              const float* __restrict__ res,
                                              void* __restrict__ outp,
                                              int M, int N, int K) {
    __shared__ _Float16 As[128 * LDA];
    __shared__ _Float16 Bs[32 * LDB];
    int tid = threadIdx.x, w = tid >> 5, lane = tid & 31, hlf = lane >> 4;
    int wm = w & 1, wn = w >> 1;          // wave tile: rows wm*64..+64, cols wn*64..+64
    int m0 = blockIdx.y * 128, n0 = blockIdx.x * 256;

    v8f acc[4][4];
#pragma unroll
    for (int mt = 0; mt < 4; ++mt)
#pragma unroll
        for (int nt = 0; nt < 4; ++nt)
#pragma unroll
            for (int v = 0; v < 8; ++v) acc[mt][nt][v] = 0.f;

    int nk = K >> 5;
    for (int kt = 0; kt < nk; ++kt) {
        __syncthreads();
        // stage A: 128x32 halves (1024 x 8B), B: 32x256 halves (2048 x 8B)
        for (int i = tid; i < 1024; i += 256) {
            int ra = i >> 3, sa = (i & 7) << 2;
            cp8(&As[ra * LDA + sa], &A[(size_t)(m0 + ra) * K + kt * 32 + sa]);
        }
        for (int i = tid; i < 2048; i += 256) {
            int rb = i >> 6, sb = (i & 63) << 2;
            cp8(&Bs[rb * LDB + sb], &B[(size_t)(kt * 32 + rb) * N + n0 + sb]);
        }
        if (kt + 1 < nk) {  // global_prefetch_b8 next tiles
            if (tid < 128)
                __builtin_prefetch(&A[(size_t)(m0 + tid) * K + (kt + 1) * 32], 0, 1);
            else if (tid < 160)
                __builtin_prefetch(&B[(size_t)((kt + 1) * 32 + (tid - 128)) * N + n0], 0, 1);
            else if (tid < 192)
                __builtin_prefetch(&B[(size_t)((kt + 1) * 32 + (tid - 160)) * N + n0 + 128], 0, 1);
        }
        async_fence();
        __syncthreads();

        v16h af[4];
#pragma unroll
        for (int mt = 0; mt < 4; ++mt)
            af[mt] = load_a_frag(As, LDA, wm * 64 + mt * 16, 0, lane);
#pragma unroll
        for (int nt = 0; nt < 4; ++nt) {
            v16h bf = load_b_frag(Bs, LDB, wn * 64 + nt * 16, lane);
#pragma unroll
            for (int mt = 0; mt < 4; ++mt)
                acc[mt][nt] = wmma_f16(af[mt], bf, acc[mt][nt]);
        }
    }

    int n = lane & 15;
#pragma unroll
    for (int nt = 0; nt < 4; ++nt) {
        int col = n0 + wn * 64 + nt * 16 + n;
        float bv = bias[col];
#pragma unroll
        for (int mt = 0; mt < 4; ++mt)
#pragma unroll
            for (int v = 0; v < 8; ++v) {
                int row = m0 + wm * 64 + mt * 16 + v + 8 * hlf;
                float val = acc[mt][nt][v] + bv;
                if (MODE == 0) {
                    ((_Float16*)outp)[(size_t)row * N + col] = (_Float16)val;
                } else if (MODE == 1) {
                    float ge = 0.5f * val * (1.f + erff(val * 0.70710678118654752f));
                    ((_Float16*)outp)[(size_t)row * N + col] = (_Float16)ge;
                } else {
                    ((float*)outp)[(size_t)row * N + col] = val + res[(size_t)row * N + col];
                }
            }
    }
}

// ---------------- Flash attention: 8 waves x 16 queries, 32-key chunks ------------
#define LDK 68
__global__ __launch_bounds__(256) void attn_k(const _Float16* __restrict__ qkv,
                                              _Float16* __restrict__ ctx16) {
    __shared__ _Float16 Ks[32 * LDK];
    __shared__ _Float16 Vs[32 * LDK];
    __shared__ float    Ss[8][16][32];
    __shared__ _Float16 Ps[8][16 * 34];
    __shared__ float    mst[8][16], lst[8][16], alA[8][16];

    int tid = threadIdx.x, w = tid >> 5, lane = tid & 31, hlf = lane >> 4;
    int bh = blockIdx.y, b = bh >> 4, h = bh & 15;
    int qtok0 = b * L_SEQ + blockIdx.x * 128 + w * 16;

    const _Float16* Qg = qkv + (size_t)qtok0 * (3 * D_MODEL) + h * HDIM;
    v16h aQ0 = load_a_frag(Qg, 3 * D_MODEL, 0, 0, lane);
    v16h aQ1 = load_a_frag(Qg, 3 * D_MODEL, 0, 32, lane);

    v8f ctx[4];
#pragma unroll
    for (int nt = 0; nt < 4; ++nt)
#pragma unroll
        for (int v = 0; v < 8; ++v) ctx[nt][v] = 0.f;

    if (lane < 16) { mst[w][lane] = -1e30f; lst[w][lane] = 0.f; }

    for (int kc = 0; kc < L_SEQ / 32; ++kc) {
        __syncthreads();
        int ktok0 = b * L_SEQ + kc * 32;
        for (int i = tid; i < 512; i += 256) {   // 32 keys x 64 dims, 8B chunks
            int key = i >> 4, seg = (i & 15) << 2;
            size_t gk = (size_t)(ktok0 + key) * (3 * D_MODEL) + D_MODEL + h * HDIM + seg;
            cp8(&Ks[key * LDK + seg], &qkv[gk]);
            cp8(&Vs[key * LDK + seg], &qkv[gk + D_MODEL]);
        }
        if (kc + 1 < L_SEQ / 32 && tid < 64) {  // prefetch next K/V rows
            int key = tid & 31;
            size_t gp = (size_t)(ktok0 + 32 + key) * (3 * D_MODEL) + D_MODEL +
                        ((tid < 32) ? 0 : D_MODEL) + h * HDIM;
            __builtin_prefetch(&qkv[gp], 0, 1);
        }
        async_fence();
        __syncthreads();

        // S(16x32) = Q(16x64) @ K^T(64x32), scaled by 1/sqrt(64)
        int n = lane & 15;
#pragma unroll
        for (int nt = 0; nt < 2; ++nt) {
            v8f s;
#pragma unroll
            for (int v = 0; v < 8; ++v) s[v] = 0.f;
            v16h b0 = load_bT_frag(Ks, LDK, nt * 16, 0, lane);
            s = wmma_f16(aQ0, b0, s);
            v16h b1 = load_bT_frag(Ks, LDK, nt * 16, 32, lane);
            s = wmma_f16(aQ1, b1, s);
#pragma unroll
            for (int v = 0; v < 8; ++v)
                Ss[w][v + 8 * hlf][nt * 16 + n] = s[v] * 0.125f;
        }

        // online softmax: lane r owns row r (same-wave LDS ops are in-order)
        if (lane < 16) {
            float mo = mst[w][lane], cm = mo;
            for (int j = 0; j < 32; ++j) cm = fmaxf(cm, Ss[w][lane][j]);
            float alpha = __expf(mo - cm), sum = 0.f;
            for (int j = 0; j < 32; ++j) {
                float p = __expf(Ss[w][lane][j] - cm);
                Ps[w][lane * 34 + j] = (_Float16)p;
                sum += p;
            }
            lst[w][lane] = lst[w][lane] * alpha + sum;
            mst[w][lane] = cm;
            alA[w][lane] = alpha;
        }

        // ctx = ctx*alpha + P(16x32) @ V(32x64)
        v16h aP = load_a_frag(Ps[w], 34, 0, 0, lane);
#pragma unroll
        for (int nt = 0; nt < 4; ++nt) {
#pragma unroll
            for (int v = 0; v < 8; ++v) ctx[nt][v] *= alA[w][v + 8 * hlf];
            v16h bV = load_b_frag(Vs, LDK, nt * 16, lane);
            ctx[nt] = wmma_f16(aP, bV, ctx[nt]);
        }
    }

    if (lane < 16) alA[w][lane] = 1.0f / lst[w][lane];
    int n = lane & 15;
#pragma unroll
    for (int nt = 0; nt < 4; ++nt)
#pragma unroll
        for (int v = 0; v < 8; ++v) {
            int m = v + 8 * hlf;
            ctx16[(size_t)(qtok0 + m) * D_MODEL + h * HDIM + nt * 16 + n] =
                (_Float16)(ctx[nt][v] * alA[w][m]);
        }
}

extern "C" void kernel_launch(void* const* d_in, const int* in_sizes, int n_in,
                              void* d_out, int out_size, void* d_ws, size_t ws_size,
                              hipStream_t stream) {
    (void)in_sizes; (void)n_in; (void)out_size; (void)ws_size;
    const float* x     = (const float*)d_in[0];
    const float* qkv_w = (const float*)d_in[1];
    const float* qkv_b = (const float*)d_in[2];
    const float* out_w = (const float*)d_in[3];
    const float* out_b = (const float*)d_in[4];
    const float* ff1_w = (const float*)d_in[5];
    const float* ff1_b = (const float*)d_in[6];
    const float* ff2_w = (const float*)d_in[7];
    const float* ff2_b = (const float*)d_in[8];
    const float* ln1_g = (const float*)d_in[9];
    const float* ln1_b = (const float*)d_in[10];
    const float* ln2_g = (const float*)d_in[11];
    const float* ln2_b = (const float*)d_in[12];
    float* out = (float*)d_out;

    char* ws = (char*)d_ws;
    _Float16* wqkv  = (_Float16*)(ws + 0);            // 1024*3072 f16 (6 MB)
    _Float16* wout  = (_Float16*)(ws + 6291456);      // 1024*1024 f16 (2 MB)
    _Float16* wff1  = (_Float16*)(ws + 8388608);      // 1024*2048 f16 (4 MB)
    _Float16* wff2  = (_Float16*)(ws + 12582912);     // 2048*1024 f16 (4 MB)
    _Float16* h16   = (_Float16*)(ws + 16777216);     // 8192*1024 f16 (16 MB)
    _Float16* qkv16 = (_Float16*)(ws + 33554432);     // 8192*3072 f16 (48 MB)
    _Float16* ffh16 = qkv16;                          // reuse after attention
    _Float16* ctx16 = (_Float16*)(ws + 83886080);     // 8192*1024 f16 (16 MB)
    float*    x2    = (float*)(ws + 100663296);       // 8192*1024 f32 (32 MB) -> 128 MB

    cvt_k<<<(3145728 + 255) / 256, 256, 0, stream>>>(qkv_w, wqkv, 3145728);
    cvt_k<<<(1048576 + 255) / 256, 256, 0, stream>>>(out_w, wout, 1048576);
    cvt_k<<<(2097152 + 255) / 256, 256, 0, stream>>>(ff1_w, wff1, 2097152);
    cvt_k<<<(2097152 + 255) / 256, 256, 0, stream>>>(ff2_w, wff2, 2097152);

    ln_k<<<NTOK, 256, 0, stream>>>(x, ln1_g, ln1_b, h16);
    gemm_k<0><<<dim3(3072 / 256, NTOK / 128), 256, 0, stream>>>(
        h16, wqkv, qkv_b, nullptr, qkv16, NTOK, 3072, 1024);
    attn_k<<<dim3(L_SEQ / 128, 4 * NHEAD), 256, 0, stream>>>(qkv16, ctx16);
    gemm_k<2><<<dim3(1024 / 256, NTOK / 128), 256, 0, stream>>>(
        ctx16, wout, out_b, x, x2, NTOK, 1024, 1024);
    ln_k<<<NTOK, 256, 0, stream>>>(x2, ln2_g, ln2_b, h16);
    gemm_k<1><<<dim3(2048 / 256, NTOK / 128), 256, 0, stream>>>(
        h16, wff1, ff1_b, nullptr, ffh16, NTOK, 2048, 1024);
    gemm_k<2><<<dim3(1024 / 256, NTOK / 128), 256, 0, stream>>>(
        ffh16, wff2, ff2_b, x2, out, NTOK, 1024, 2048);
}
